// FractalAttention_15083925144011
// MI455X (gfx1250) — matrix operand order repaired and checked by reading, not compiled, not run on hardware
//
#include <hip/hip_runtime.h>

// ---------------------------------------------------------------------------
// FractalAttention for MI455X (gfx1250, wave32, WMMA f32_16x16x32_f16,
// TDM tensor_load_to_lds double-buffered staging for K tiles and weights)
// ---------------------------------------------------------------------------

typedef __attribute__((ext_vector_type(16))) _Float16 v16h;
typedef __attribute__((ext_vector_type(8)))  _Float16 v8h;
typedef __attribute__((ext_vector_type(4)))  _Float16 v4h;
typedef __attribute__((ext_vector_type(8)))  float    v8f;
typedef __attribute__((ext_vector_type(4)))  unsigned int v4u;
typedef __attribute__((ext_vector_type(8)))  int      v8i;
typedef __attribute__((ext_vector_type(4)))  int      v4i;

#define SEQ   4096
#define DIM   1024
#define NHEAD 16
#define HDIM  64

#if __has_include(<hip/amd_detail/amd_gfx1250_TDM.h>)
#define FA_TDM_6ARG 1
#else
#define FA_TDM_6ARG 0
#endif

__device__ __forceinline__ v8f wmma_f16(v16h a, v16h b, v8f c) {
  return __builtin_amdgcn_wmma_f32_16x16x32_f16(false, a, false, b, (short)0, c,
                                                false, false);
}

// A-operand (16x32 f16), row-major source [row][k]:
// lanes 0-15 : row = lane,    K = k0+{0..7}, k0+{16..23}
// lanes 16-31: row = lane-16, K = k0+{8..15}, k0+{24..31}
__device__ __forceinline__ v16h load_a16(const _Float16* base, int ld, int row0, int k0) {
  const int lane = (int)(threadIdx.x & 31u);
  const _Float16* p = base + (size_t)(row0 + (lane & 15)) * (size_t)ld
                           + (size_t)(k0 + ((lane >> 4) << 3));
  union { v16h v; v8h h[2]; } u;
  u.h[0] = *(const v8h*)(p);
  u.h[1] = *(const v8h*)(p + 16);
  return u.v;
}

// B-operand (32x16 f16) from row-major source [n][k]:
// lanes 0-15 : col = lane,    K = k0+{0..7}, k0+{16..23}
// lanes 16-31: col = lane-16, K = k0+{8..15}, k0+{24..31}
__device__ __forceinline__ v16h load_b16(const _Float16* base, int ld, int n0, int k0) {
  const int lane = (int)(threadIdx.x & 31u);
  const _Float16* p = base + (size_t)(n0 + (lane & 15)) * (size_t)ld
                           + (size_t)(k0 + ((lane >> 4) << 3));
  union { v16h v; v8h h[2]; } u;
  u.h[0] = *(const v8h*)(p);
  u.h[1] = *(const v8h*)(p + 16);
  return u.v;
}

// TDM: DMA a 2-D f16 tile (tile_w contiguous elems x tile_h rows, row stride
// `stride` elems) from global into LDS. D# groups 2/3 zeroed (<=2D tensor).
__device__ __forceinline__ void tdm_load_tile2d(unsigned lds_addr,
                                                const _Float16* gptr,
                                                unsigned tile_w, unsigned tile_h,
                                                unsigned stride,
                                                unsigned dim0, unsigned dim1) {
  const unsigned long long ga = (unsigned long long)(uintptr_t)gptr;
  v4u g0;
  g0.x = 1u;                                                // count=1, user D#
  g0.y = lds_addr;                                          // lds_addr[31:0]
  g0.z = (unsigned)(ga & 0xffffffffu);                      // global_addr[31:0]
  g0.w = (unsigned)((ga >> 32) & 0x01ffffffu) | (2u << 30); // ga[56:32]|type=2
  v8i g1;
  g1[0] = 0x00010000;                          // wg_mask=0, data_size=1 (2 B)
  g1[1] = (int)(dim0 << 16);                   // tensor_dim0[15:0]
  g1[2] = (int)((dim0 >> 16) | (dim1 << 16));  // dim0[31:16] | dim1[15:0]
  g1[3] = (int)((dim1 >> 16) | (tile_w << 16));// dim1[31:16] | tile_dim0
  g1[4] = (int)tile_h;                         // tile_dim1 (tile_dim2 = 0)
  g1[5] = (int)stride;                         // tensor_dim0_stride[31:0]
  g1[6] = 0;
  g1[7] = 0;
  const v4i z4 = (v4i){0, 0, 0, 0};
#if FA_TDM_6ARG
  const v8i z8 = (v8i){0, 0, 0, 0, 0, 0, 0, 0};
  __builtin_amdgcn_tensor_load_to_lds(g0, g1, z4, z4, z8, 0);
#else
  __builtin_amdgcn_tensor_load_to_lds(g0, g1, z4, z4, 0);
#endif
}

// ---------------------------------------------------------------------------
// 1) fp32 -> fp16 conversion (vectorized x4)
// ---------------------------------------------------------------------------
__global__ void fa_cvt_f32_f16(const float* __restrict__ src,
                               _Float16* __restrict__ dst, int n4) {
  int i = blockIdx.x * blockDim.x + threadIdx.x;
  if (i < n4) {
    float4 v = ((const float4*)src)[i];
    v4h o;
    o.x = (_Float16)v.x; o.y = (_Float16)v.y;
    o.z = (_Float16)v.z; o.w = (_Float16)v.w;
    ((v4h*)dst)[i] = o;
  }
}

// ---------------------------------------------------------------------------
// 2) Fused QKV projection. All 8 waves of a block share the same 64-column
//    weight tiles, so Wq/Wk/Wv 64x32 tiles are DMA'd by TDM into
//    double-buffered LDS (3 descriptors per k-step) while waves compute the
//    previous k-step. The unique per-wave A (x) tile is software-pipelined
//    through registers. One barrier per k-step.
// ---------------------------------------------------------------------------
__global__ __launch_bounds__(256) void fa_qkv_proj(
    const _Float16* __restrict__ xh,
    const _Float16* __restrict__ Wqh, const _Float16* __restrict__ Wkh,
    const _Float16* __restrict__ Wvh,
    const float* __restrict__ bq, const float* __restrict__ bk,
    const float* __restrict__ bv,
    _Float16* __restrict__ Qh, _Float16* __restrict__ Kh,
    _Float16* __restrict__ Vh) {
  __shared__ __attribute__((aligned(16))) _Float16 Wt[2][3][64 * 32];

  const int lane = (int)(threadIdx.x & 31u);
  const int wave = (int)(threadIdx.x >> 5u);
  const int s0 = blockIdx.x * 128 + wave * 16;
  const int n0 = blockIdx.y * 64;

  v8f aq[4], ak[4], av[4];
#pragma unroll
  for (int nt = 0; nt < 4; ++nt) {
    aq[nt] = (v8f){0,0,0,0,0,0,0,0};
    ak[nt] = (v8f){0,0,0,0,0,0,0,0};
    av[nt] = (v8f){0,0,0,0,0,0,0,0};
  }

  const size_t wrow = (size_t)n0 * DIM;
  // prologue: DMA weight tiles for k0 = 0 into buffer 0
  if (wave == 0) {
    tdm_load_tile2d((unsigned)(uintptr_t)&Wt[0][0][0], Wqh + wrow, 32, 64, DIM, DIM, DIM);
    tdm_load_tile2d((unsigned)(uintptr_t)&Wt[0][1][0], Wkh + wrow, 32, 64, DIM, DIM, DIM);
    tdm_load_tile2d((unsigned)(uintptr_t)&Wt[0][2][0], Wvh + wrow, 32, 64, DIM, DIM, DIM);
  }
  v16h a_cur = load_a16(xh, DIM, s0, 0);

  int buf = 0;
  for (int k0 = 0; k0 < DIM; k0 += 32, buf ^= 1) {
    if (wave == 0) __builtin_amdgcn_s_wait_tensorcnt(0);  // Wt[buf] DMA done
    __syncthreads();  // publish Wt[buf]; all waves done reading Wt[buf^1]

    const bool have_next = (k0 + 32 < DIM);
    if (have_next && wave == 0) {
      const size_t g = wrow + (size_t)(k0 + 32);
      tdm_load_tile2d((unsigned)(uintptr_t)&Wt[buf ^ 1][0][0], Wqh + g, 32, 64, DIM, DIM, DIM);
      tdm_load_tile2d((unsigned)(uintptr_t)&Wt[buf ^ 1][1][0], Wkh + g, 32, 64, DIM, DIM, DIM);
      tdm_load_tile2d((unsigned)(uintptr_t)&Wt[buf ^ 1][2][0], Wvh + g, 32, 64, DIM, DIM, DIM);
    }
    // prefetch next A tile (global latency hides under this step's WMMAs)
    v16h a_next = a_cur;
    if (have_next) a_next = load_a16(xh, DIM, s0, k0 + 32);

    const _Float16* Wq_l = Wt[buf][0];
    const _Float16* Wk_l = Wt[buf][1];
    const _Float16* Wv_l = Wt[buf][2];
    v16h b[12];
#pragma unroll
    for (int nt = 0; nt < 4; ++nt) {
      b[nt * 3 + 0] = load_b16(Wq_l, 32, nt * 16, 0);
      b[nt * 3 + 1] = load_b16(Wk_l, 32, nt * 16, 0);
      b[nt * 3 + 2] = load_b16(Wv_l, 32, nt * 16, 0);
    }
#pragma unroll
    for (int nt = 0; nt < 4; ++nt) {
      aq[nt] = wmma_f16(a_cur, b[nt * 3 + 0], aq[nt]);
      ak[nt] = wmma_f16(a_cur, b[nt * 3 + 1], ak[nt]);
      av[nt] = wmma_f16(a_cur, b[nt * 3 + 2], av[nt]);
    }
    a_cur = a_next;
  }

  const int mhalf = (lane >> 4) << 3;
#pragma unroll
  for (int nt = 0; nt < 4; ++nt) {
    const int o = n0 + nt * 16 + (lane & 15);
    const int head = o >> 6, d = o & 63;
    const float bqv = bq[o], bkv = bk[o], bvv = bv[o];
    const size_t hb = (size_t)head * SEQ * HDIM + (size_t)d;
#pragma unroll
    for (int r = 0; r < 8; ++r) {
      const size_t idx = hb + (size_t)(s0 + r + mhalf) * HDIM;
      Qh[idx] = (_Float16)((aq[nt][r] + bqv) * 0.125f);  // fold D^-0.5 = 1/8
      Kh[idx] = (_Float16)(ak[nt][r] + bkv);
      Vh[idx] = (_Float16)(av[nt][r] + bvv);
    }
  }
}

// ---------------------------------------------------------------------------
// 3) Flash attention. Block = one head, 128 queries (8 waves x 16).
//    K tiles: TDM DMA into double-buffered LDS (wave 0 issues, TENSORcnt).
//    V tiles: register-staged transpose, global loads issued BEFORE the
//    chunk's compute, transposing ds_stores issued AFTER it.
// ---------------------------------------------------------------------------
__global__ __launch_bounds__(256) void fa_flash(
    const _Float16* __restrict__ Qh, const _Float16* __restrict__ Kh,
    const _Float16* __restrict__ Vh, _Float16* __restrict__ Oh) {
  __shared__ __attribute__((aligned(16))) _Float16 Kt[2][32 * 64];   // [key][d]
  __shared__ __attribute__((aligned(16))) _Float16 Vt[2][64 * 32];   // [d][key]
  __shared__ __attribute__((aligned(16))) _Float16 Pl[8][16 * 32];   // per-wave

  const int lane = (int)(threadIdx.x & 31u);
  const int wave = (int)(threadIdx.x >> 5u);
  const int head = blockIdx.y;
  const int q0 = blockIdx.x * 128 + wave * 16;

  const _Float16* Qb = Qh + (size_t)head * SEQ * HDIM;
  const _Float16* Kb = Kh + (size_t)head * SEQ * HDIM;
  const _Float16* Vb = Vh + (size_t)head * SEQ * HDIM;

  const v16h qa0 = load_a16(Qb, HDIM, q0, 0);
  const v16h qa1 = load_a16(Qb, HDIM, q0, 32);

  v8f o0 = (v8f){0,0,0,0,0,0,0,0}, o1 = o0, o2 = o0, o3 = o0;
  float mrow[8], lrow[8];
#pragma unroll
  for (int r = 0; r < 8; ++r) { mrow[r] = -1e30f; lrow[r] = 0.0f; }

  const int t = (int)threadIdx.x;
  const int vkey = t >> 3;            // 0..31
  const int vd0 = (t & 7) << 3;       // 0..56

  // ---- prologue: start filling buffer 0 (chunk 0)
  if (wave == 0)
    tdm_load_tile2d((unsigned)(uintptr_t)&Kt[0][0], Kb, 64, 32, 64, 64, SEQ);
  {
    v8h vv = *(const v8h*)(Vb + (size_t)vkey * HDIM + vd0);
#pragma unroll
    for (int i = 0; i < 8; ++i) Vt[0][(vd0 + i) * 32 + vkey] = vv[i];
  }

  int buf = 0;
  for (int kc = 0; kc < SEQ; kc += 32, buf ^= 1) {
    if (wave == 0) __builtin_amdgcn_s_wait_tensorcnt(0);  // K[buf] DMA done
    __syncthreads();  // publish K/V[buf]; all waves done with K/V[buf^1]

    // ---- kick off next chunk: TDM for K, global loads for V (stores later)
    const bool have_next = (kc + 32 < SEQ);
    v8h vstage = (v8h){0, 0, 0, 0, 0, 0, 0, 0};
    if (have_next) {
      if (wave == 0)
        tdm_load_tile2d((unsigned)(uintptr_t)&Kt[buf ^ 1][0],
                        Kb + (size_t)(kc + 32) * HDIM, 64, 32, 64, 64, SEQ);
      vstage = *(const v8h*)(Vb + (size_t)(kc + 32 + vkey) * HDIM + vd0);
    }

    const _Float16* Kc = Kt[buf];
    const _Float16* Vc = Vt[buf];

    // ---- S(16x32) = Q(16x64) x K^T : batch 4 LDS B-loads, then 4 WMMAs
    v16h kb0 = load_b16(Kc, 64, 0, 0);
    v16h kb1 = load_b16(Kc, 64, 0, 32);
    v16h kb2 = load_b16(Kc, 64, 16, 0);
    v16h kb3 = load_b16(Kc, 64, 16, 32);
    v8f s0 = (v8f){0,0,0,0,0,0,0,0}, s1 = s0;
    s0 = wmma_f16(qa0, kb0, s0);
    s0 = wmma_f16(qa1, kb1, s0);
    s1 = wmma_f16(qa0, kb2, s1);
    s1 = wmma_f16(qa1, kb3, s1);

    // ---- online softmax (rows at fixed VGPR index per 16-lane half)
    _Float16* Pw = Pl[wave];
    const int mhalf = (lane >> 4) << 3;
    const int n = lane & 15;
#pragma unroll
    for (int r = 0; r < 8; ++r) {
      float mx = fmaxf(s0[r], s1[r]);
      mx = fmaxf(mx, __shfl_xor(mx, 1, 32));
      mx = fmaxf(mx, __shfl_xor(mx, 2, 32));
      mx = fmaxf(mx, __shfl_xor(mx, 4, 32));
      mx = fmaxf(mx, __shfl_xor(mx, 8, 32));
      const float mnew = fmaxf(mrow[r], mx);
      const float corr = __expf(mrow[r] - mnew);
      const float p0 = __expf(s0[r] - mnew);
      const float p1 = __expf(s1[r] - mnew);
      float rs = p0 + p1;
      rs += __shfl_xor(rs, 1, 32);
      rs += __shfl_xor(rs, 2, 32);
      rs += __shfl_xor(rs, 4, 32);
      rs += __shfl_xor(rs, 8, 32);
      lrow[r] = lrow[r] * corr + rs;
      mrow[r] = mnew;
      o0[r] *= corr; o1[r] *= corr; o2[r] *= corr; o3[r] *= corr;
      const int m = r + mhalf;
      Pw[m * 32 + n]      = (_Float16)p0;
      Pw[m * 32 + n + 16] = (_Float16)p1;
    }

    // ---- O += P(16x32) x V(32x64): batch loads, then 4 WMMAs
    {
      const v16h pa = load_a16(Pw, 32, 0, 0);
      v16h vb0 = load_b16(Vc, 32, 0, 0);
      v16h vb1 = load_b16(Vc, 32, 16, 0);
      v16h vb2 = load_b16(Vc, 32, 32, 0);
      v16h vb3 = load_b16(Vc, 32, 48, 0);
      o0 = wmma_f16(pa, vb0, o0);
      o1 = wmma_f16(pa, vb1, o1);
      o2 = wmma_f16(pa, vb2, o2);
      o3 = wmma_f16(pa, vb3, o3);
    }

    // ---- finish next-chunk V staging (transposing LDS stores)
    if (have_next) {
#pragma unroll
      for (int i = 0; i < 8; ++i)
        Vt[buf ^ 1][(vd0 + i) * 32 + vkey] = vstage[i];
    }
  }

  // ---- epilogue: phi^-head / l, write [S][DIM] f16 for the final GEMM
  const float hs = __powf(1.6180339887498949f, -(float)head);
  const int mhalf = (lane >> 4) << 3;
  const int n = lane & 15;
#pragma unroll
  for (int r = 0; r < 8; ++r) {
    const float inv = hs / lrow[r];
    const size_t base = (size_t)(q0 + r + mhalf) * DIM + (size_t)head * HDIM;
    Oh[base + 0  + n] = (_Float16)(o0[r] * inv);
    Oh[base + 16 + n] = (_Float16)(o1[r] * inv);
    Oh[base + 32 + n] = (_Float16)(o2[r] * inv);
    Oh[base + 48 + n] = (_Float16)(o3[r] * inv);
  }
}

// ---------------------------------------------------------------------------
// 4) Output projection: out = Oh @ Wo^T + bo (fp32 output).
//    Wo tiles staged by TDM (double-buffered), A pipelined through registers.
// ---------------------------------------------------------------------------
__global__ __launch_bounds__(256) void fa_out_proj(
    const _Float16* __restrict__ Oh, const _Float16* __restrict__ Woh,
    const float* __restrict__ bo, float* __restrict__ out) {
  __shared__ __attribute__((aligned(16))) _Float16 Wt[2][64 * 32];

  const int lane = (int)(threadIdx.x & 31u);
  const int wave = (int)(threadIdx.x >> 5u);
  const int s0 = blockIdx.x * 128 + wave * 16;
  const int n0 = blockIdx.y * 64;

  v8f acc[4];
#pragma unroll
  for (int nt = 0; nt < 4; ++nt) acc[nt] = (v8f){0,0,0,0,0,0,0,0};

  const size_t wrow = (size_t)n0 * DIM;
  if (wave == 0)
    tdm_load_tile2d((unsigned)(uintptr_t)&Wt[0][0], Woh + wrow, 32, 64, DIM, DIM, DIM);
  v16h a_cur = load_a16(Oh, DIM, s0, 0);

  int buf = 0;
  for (int k0 = 0; k0 < DIM; k0 += 32, buf ^= 1) {
    if (wave == 0) __builtin_amdgcn_s_wait_tensorcnt(0);
    __syncthreads();

    const bool have_next = (k0 + 32 < DIM);
    if (have_next && wave == 0)
      tdm_load_tile2d((unsigned)(uintptr_t)&Wt[buf ^ 1][0],
                      Woh + wrow + (size_t)(k0 + 32), 32, 64, DIM, DIM, DIM);
    v16h a_next = a_cur;
    if (have_next) a_next = load_a16(Oh, DIM, s0, k0 + 32);

    const _Float16* Wl = Wt[buf];
    v16h b[4];
#pragma unroll
    for (int nt = 0; nt < 4; ++nt)
      b[nt] = load_b16(Wl, 32, nt * 16, 0);
#pragma unroll
    for (int nt = 0; nt < 4; ++nt)
      acc[nt] = wmma_f16(a_cur, b[nt], acc[nt]);
    a_cur = a_next;
  }

  const int mhalf = (lane >> 4) << 3;
#pragma unroll
  for (int nt = 0; nt < 4; ++nt) {
    const int o = n0 + nt * 16 + (lane & 15);
    const float bov = bo[o];
#pragma unroll
    for (int r = 0; r < 8; ++r)
      out[(size_t)(s0 + r + mhalf) * DIM + o] = acc[nt][r] + bov;
  }
}

// ---------------------------------------------------------------------------
// launcher
// ---------------------------------------------------------------------------
extern "C" void kernel_launch(void* const* d_in, const int* in_sizes, int n_in,
                              void* d_out, int out_size, void* d_ws, size_t ws_size,
                              hipStream_t stream) {
  const float* x  = (const float*)d_in[0];
  const float* Wq = (const float*)d_in[1];
  const float* bq = (const float*)d_in[2];
  const float* Wk = (const float*)d_in[3];
  const float* bk = (const float*)d_in[4];
  const float* Wv = (const float*)d_in[5];
  const float* bv = (const float*)d_in[6];
  const float* Wo = (const float*)d_in[7];
  const float* bo = (const float*)d_in[8];
  float* out = (float*)d_out;

  char* ws = (char*)d_ws;
  const size_t MB = 1024ull * 1024ull;
  _Float16* xh  = (_Float16*)(ws + 0 * MB);   // 8 MB
  _Float16* Wqh = (_Float16*)(ws + 8 * MB);   // 2 MB
  _Float16* Wkh = (_Float16*)(ws + 10 * MB);  // 2 MB
  _Float16* Wvh = (_Float16*)(ws + 12 * MB);  // 2 MB
  _Float16* Woh = (_Float16*)(ws + 14 * MB);  // 2 MB
  _Float16* Qh  = (_Float16*)(ws + 16 * MB);  // 8 MB  [H][S][D]
  _Float16* Kh  = (_Float16*)(ws + 24 * MB);  // 8 MB
  _Float16* Vh  = (_Float16*)(ws + 32 * MB);  // 8 MB
  _Float16* Oh  = (_Float16*)(ws + 40 * MB);  // 8 MB  [S][DIM]

  const int nx4 = SEQ * DIM / 4;
  const int nw4 = DIM * DIM / 4;
  fa_cvt_f32_f16<<<nx4 / 256, 256, 0, stream>>>(x,  xh,  nx4);
  fa_cvt_f32_f16<<<nw4 / 256, 256, 0, stream>>>(Wq, Wqh, nw4);
  fa_cvt_f32_f16<<<nw4 / 256, 256, 0, stream>>>(Wk, Wkh, nw4);
  fa_cvt_f32_f16<<<nw4 / 256, 256, 0, stream>>>(Wv, Wvh, nw4);
  fa_cvt_f32_f16<<<nw4 / 256, 256, 0, stream>>>(Wo, Woh, nw4);

  fa_qkv_proj<<<dim3(SEQ / 128, DIM / 64), 256, 0, stream>>>(
      xh, Wqh, Wkh, Wvh, bq, bk, bv, Qh, Kh, Vh);

  fa_flash<<<dim3(SEQ / 128, NHEAD), 256, 0, stream>>>(Qh, Kh, Vh, Oh);

  fa_out_proj<<<dim3(SEQ / 128, DIM / 64), 256, 0, stream>>>(Oh, Woh, bo, out);
}
